// MultiHeadAttention_43224550867548
// MI455X (gfx1250) — matrix operand order repaired and checked by reading, not compiled, not run on hardware
//
#include <hip/hip_runtime.h>

// ---------------------------------------------------------------------------
// MHA for MI455X (gfx1250): B=4, S=2048, D=1024, H=16, DH=64.
// All GEMMs via v_wmma_f32_16x16x32_bf16 (wave32). bf16 inputs, fp32 accum.
// Kernel 3 stages its A tile with the Tensor Data Mover (TENSORcnt) and its
// B tile with async global->LDS DMA (ASYNCcnt) -- both gfx1250-only paths.
// ---------------------------------------------------------------------------

typedef __bf16 v16bf __attribute__((ext_vector_type(16)));
typedef __bf16 v8bf  __attribute__((ext_vector_type(8)));
typedef float  v8f   __attribute__((ext_vector_type(8)));

union Frag16 { v16bf v; v8bf h[2]; };
union Pack8  { v8bf v; __bf16 e[8]; };

#define BATCH 4
#define SEQ   2048
#define DIM   1024
#define HEADS 16
#define DHEAD 64

static __device__ __forceinline__ v8f wmma_bf16(v16bf a, v16bf b, v8f c) {
  return __builtin_amdgcn_wmma_f32_16x16x32_bf16(false, a, false, b, (short)0, c,
                                                 false, false);
}

// gfx1250 async DMA: copy 16B/lane global -> LDS, tracked by ASYNCcnt.
static __device__ __forceinline__ void async_copy_b128(unsigned lds_off,
                                                       const void* gptr) {
  asm volatile("global_load_async_to_lds_b128 %0, %1, off"
               :: "v"(lds_off), "v"((unsigned long long)(uintptr_t)gptr)
               : "memory");
}
static __device__ __forceinline__ void wait_asynccnt0() {
  asm volatile("s_wait_asynccnt 0x0" ::: "memory");
}

#if __has_builtin(__builtin_amdgcn_tensor_load_to_lds)
#define HAVE_TDM 1
typedef unsigned int u32x4 __attribute__((ext_vector_type(4)));
typedef int          i32x8 __attribute__((ext_vector_type(8)));
typedef int          i32x4 __attribute__((ext_vector_type(4)));
#endif

// ---------------------------------------------------------------------------
// Kernel 1: qkv = x @ w_qkv ; scatter to Q[b,h,s,dh] (*1/8), K[b,h,s,dh],
// Vt[b,h,dh,s] (transposed) in bf16.  M=8192, N=3072, K=1024.
// 4 waves; block tile 64x64; K-step 64 -> 8 WMMAs per wave per barrier pair.
// ---------------------------------------------------------------------------
__global__ __launch_bounds__(128) void qkv_proj_kernel(
    const float* __restrict__ x, const float* __restrict__ w,
    __bf16* __restrict__ Qw, __bf16* __restrict__ Kw, __bf16* __restrict__ Vtw)
{
  __shared__ __bf16 At[64 * 64];  // [row][k], row stride 128B
  __shared__ __bf16 Bt[64 * 64];  // [n][k] transposed weight tile

  const int tid  = threadIdx.x;
  const int lane = tid & 31, wid = tid >> 5;
  const int hl = lane >> 4, l16 = lane & 15;

  const int nTilesN = (3 * DIM) / 64;            // 48
  const int m0 = (blockIdx.x / nTilesN) * 64;
  const int n0 = (blockIdx.x % nTilesN) * 64;
  const int wm = (wid >> 1) * 32, wn = (wid & 1) * 32;

  v8f acc[2][2] = {};

  for (int k0 = 0; k0 < DIM; k0 += 64) {
    { // stage A (fp32 -> bf16): thread t -> row t/2, 32-elem half t&1
      const int row = tid >> 1, half = tid & 1;
      const float* src = x + (size_t)(m0 + row) * DIM + k0 + 32 * half;
      __bf16* dst = At + row * 64 + 32 * half;
      #pragma unroll
      for (int i = 0; i < 32; ++i) dst[i] = (__bf16)src[i];
      if (k0 + 64 < DIM) __builtin_prefetch(src + 64, 0, 1);  // global_prefetch
    }
    { // stage B transposed: thread t -> column n=t&63, 32-k group
      const int n = tid & 63, kk = (tid >> 6) * 32;
      const float* src = w + (size_t)(k0 + kk) * (3 * DIM) + n0 + n;
      __bf16* dst = Bt + n * 64 + kk;
      #pragma unroll
      for (int i = 0; i < 32; ++i) dst[i] = (__bf16)src[(size_t)i * (3 * DIM)];
    }
    __syncthreads();

    Frag16 a[2][2], b[2][2];
    #pragma unroll
    for (int mi = 0; mi < 2; ++mi)
      #pragma unroll
      for (int kc = 0; kc < 2; ++kc) {
        const __bf16* rowp = At + (wm + 16 * mi + l16) * 64 + 32 * kc;
        a[mi][kc].v = *(const v16bf*)(rowp + 16 * hl);
      }
    #pragma unroll
    for (int ni = 0; ni < 2; ++ni)
      #pragma unroll
      for (int kc = 0; kc < 2; ++kc)
        b[ni][kc].v =
            *(const v16bf*)(Bt + (wn + 16 * ni + l16) * 64 + 32 * kc + 16 * hl);
    #pragma unroll
    for (int mi = 0; mi < 2; ++mi)
      #pragma unroll
      for (int ni = 0; ni < 2; ++ni)
        #pragma unroll
        for (int kc = 0; kc < 2; ++kc)
          acc[mi][ni] = wmma_bf16(a[mi][kc].v, b[ni][kc].v, acc[mi][ni]);
    __syncthreads();
  }

  // Epilogue. which/h/b/s0 are uniform per 16x16 subtile -> force scalar.
  const int wm_s = __builtin_amdgcn_readfirstlane(wm);
  const int wn_s = __builtin_amdgcn_readfirstlane(wn);
  #pragma unroll
  for (int mi = 0; mi < 2; ++mi)
    #pragma unroll
    for (int ni = 0; ni < 2; ++ni) {
      const int ms = m0 + wm_s + 16 * mi;          // scalar
      const int ns = n0 + wn_s + 16 * ni;          // scalar
      const int which = ns >> 10, h = (ns & 1023) >> 6, dh0 = ns & 63;
      const int bb = ms >> 11, s0 = ms & (SEQ - 1);
      const size_t bh = (size_t)bb * HEADS + h;
      if (which == 2) {  // V transposed: 8 rows contiguous in s -> one b128
        Pack8 pk;
        #pragma unroll
        for (int vr = 0; vr < 8; ++vr) pk.e[vr] = (__bf16)acc[mi][ni][vr];
        *(v8bf*)(Vtw + (bh * DHEAD + dh0 + l16) * SEQ + s0 + 8 * hl) = pk.v;
      } else if (which == 0) {
        __bf16* p = Qw + (bh * SEQ + s0 + 8 * hl) * DHEAD + dh0 + l16;
        #pragma unroll
        for (int vr = 0; vr < 8; ++vr)
          p[vr * DHEAD] = (__bf16)(acc[mi][ni][vr] * 0.125f);  // 1/sqrt(64)
      } else {
        __bf16* p = Kw + (bh * SEQ + s0 + 8 * hl) * DHEAD + dh0 + l16;
        #pragma unroll
        for (int vr = 0; vr < 8; ++vr) p[vr * DHEAD] = (__bf16)acc[mi][ni][vr];
      }
    }
}

// ---------------------------------------------------------------------------
// Kernel 2: causal flash attention. One wave per 32-query tile (2 q-subtiles),
// key blocks of 32, online softmax via 16-lane-half shuffles, P converted
// C-layout -> A-layout through a per-wave LDS tile. 16 WMMAs per key block.
// ---------------------------------------------------------------------------
__global__ __launch_bounds__(128) void attn_kernel(
    const __bf16* __restrict__ Qw, const __bf16* __restrict__ Kw,
    const __bf16* __restrict__ Vtw, __bf16* __restrict__ Ow)
{
  __shared__ __bf16 Pt[4][2][16 * 32];   // per-wave, per-q-subtile P tile

  const int tid = threadIdx.x, lane = tid & 31, wid = tid >> 5;
  const int hl = lane >> 4, l16 = lane & 15;

  const int tile = blockIdx.x * 4 + wid;       // 0..4095
  const int qb = tile & (SEQ / 32 - 1);        // 0..63
  const int bh = tile >> 6;                    // b*HEADS + h
  const int q0 = qb * 32;

  const __bf16* Qp = Qw + ((size_t)bh * SEQ + q0) * DHEAD;
  const __bf16* Kp = Kw + (size_t)bh * SEQ * DHEAD;
  const __bf16* Vp = Vtw + (size_t)bh * DHEAD * SEQ;

  // Q A-fragments (2 q-subtiles x 2 contraction windows), kept live.
  Frag16 aq[2][2];
  #pragma unroll
  for (int qm = 0; qm < 2; ++qm)
    #pragma unroll
    for (int ks = 0; ks < 2; ++ks) {
      const __bf16* rowp = Qp + (16 * qm + l16) * DHEAD + 32 * ks;
      aq[qm][ks].v = *(const v16bf*)(rowp + 16 * hl);
    }

  v8f o[2][4] = {};
  float mx[2][8], li[2][8];
  #pragma unroll
  for (int qm = 0; qm < 2; ++qm)
    #pragma unroll
    for (int r = 0; r < 8; ++r) { mx[qm][r] = -1e30f; li[qm][r] = 0.f; }

  const int jmax = qb;  // (q0+31)/32
  for (int j = 0; j <= jmax; ++j) {
    const int kbase = j * 32;

    // scores: both q-subtiles share each K fragment
    v8f s[2][2] = {};
    #pragma unroll
    for (int sub = 0; sub < 2; ++sub)
      #pragma unroll
      for (int ks = 0; ks < 2; ++ks) {
        const v16bf bk = *(const v16bf*)(
            Kp + (size_t)(kbase + 16 * sub + l16) * DHEAD + 32 * ks + 16 * hl);
        #pragma unroll
        for (int qm = 0; qm < 2; ++qm)
          s[qm][sub] = wmma_bf16(aq[qm][ks].v, bk, s[qm][sub]);
      }

    if (j == jmax) {  // wave-uniform: causal mask on the diagonal block
      #pragma unroll
      for (int qm = 0; qm < 2; ++qm)
        #pragma unroll
        for (int sub = 0; sub < 2; ++sub) {
          const int key = kbase + 16 * sub + l16;
          #pragma unroll
          for (int vr = 0; vr < 8; ++vr)
            if (key > q0 + 16 * qm + vr + 8 * hl) s[qm][sub][vr] = -1e30f;
        }
    }

    // online softmax (row = vr + 8*hl lives across one 16-lane half)
    #pragma unroll
    for (int qm = 0; qm < 2; ++qm) {
      __bf16* Pw = Pt[wid][qm];
      float scale[8];
      #pragma unroll
      for (int vr = 0; vr < 8; ++vr) {
        float rm = fmaxf(s[qm][0][vr], s[qm][1][vr]);
        #pragma unroll
        for (int m = 1; m < 16; m <<= 1) rm = fmaxf(rm, __shfl_xor(rm, m, 32));
        const float mnew = fmaxf(mx[qm][vr], rm);
        const float sc = __expf(mx[qm][vr] - mnew);
        const float p0 = __expf(s[qm][0][vr] - mnew);
        const float p1 = __expf(s[qm][1][vr] - mnew);
        float rs = p0 + p1;
        #pragma unroll
        for (int m = 1; m < 16; m <<= 1) rs += __shfl_xor(rs, m, 32);
        li[qm][vr] = li[qm][vr] * sc + rs;
        mx[qm][vr] = mnew;
        scale[vr] = sc;
        const int prow = vr + 8 * hl;
        Pw[prow * 32 + l16] = (__bf16)p0;
        Pw[prow * 32 + 16 + l16] = (__bf16)p1;
      }
      #pragma unroll
      for (int t = 0; t < 4; ++t)
        #pragma unroll
        for (int vr = 0; vr < 8; ++vr) o[qm][t][vr] *= scale[vr];
    }

    // reload P in A-layout (wave-local LDS, ordered by DScnt)
    Frag16 ap[2];
    #pragma unroll
    for (int qm = 0; qm < 2; ++qm) {
      const __bf16* rowp = Pt[wid][qm] + l16 * 32;
      ap[qm].v = *(const v16bf*)(rowp + 16 * hl);
    }
    // O += P(16x32) . V(32x64): V^T rows contiguous in s, shared across qm
    #pragma unroll
    for (int t = 0; t < 4; ++t) {
      const v16bf bv = *(const v16bf*)(
          Vp + (size_t)(16 * t + l16) * SEQ + kbase + 16 * hl);
      #pragma unroll
      for (int qm = 0; qm < 2; ++qm)
        o[qm][t] = wmma_bf16(ap[qm].v, bv, o[qm][t]);
    }
  }

  // epilogue: Ow[b][s][h][dh] bf16
  const int b = bh >> 4, h = bh & (HEADS - 1);
  #pragma unroll
  for (int qm = 0; qm < 2; ++qm)
    #pragma unroll
    for (int t = 0; t < 4; ++t)
      #pragma unroll
      for (int vr = 0; vr < 8; ++vr) {
        const int q = q0 + 16 * qm + vr + 8 * hl;
        const int dh = 16 * t + l16;
        const float val = o[qm][t][vr] / li[qm][vr];
        Ow[(((size_t)b * SEQ + q) * HEADS + h) * DHEAD + dh] = (__bf16)val;
      }
}

// ---------------------------------------------------------------------------
// Kernel 3: out = attn(bf16) @ w_out(fp32->bf16), fp32 result. M=8192, N=K=1024.
// A tile: one TDM tensor_load_to_lds per K-step (64x64 bf16 tile, layout
// matches At exactly). B tile: contiguous async DMA of raw fp32 into LDS,
// then a conflict-free LDS convert/transpose pass (replaces stride-4KB
// global loads with contiguous DMA).
// ---------------------------------------------------------------------------
__global__ __launch_bounds__(128) void out_proj_kernel(
    const __bf16* __restrict__ a_in, const float* __restrict__ w,
    float* __restrict__ out)
{
  __shared__ __bf16 At[64 * 64];   // 8 KB
  __shared__ __bf16 Bt[64 * 64];   // 8 KB  [n][k] bf16
  __shared__ float  Bf[64 * 64];   // 16 KB raw fp32 [k][n] staging

  const int tid = threadIdx.x;
  const int lane = tid & 31, wid = tid >> 5;
  const int hl = lane >> 4, l16 = lane & 15;

  const int nTilesN = DIM / 64;                  // 16
  const int m0 = (blockIdx.x / nTilesN) * 64;
  const int n0 = (blockIdx.x % nTilesN) * 64;
  const int wm = (wid >> 1) * 32, wn = (wid & 1) * 32;

  v8f acc[2][2] = {};

  for (int k0 = 0; k0 < DIM; k0 += 64) {
#if HAVE_TDM
    if (wid == 0) {  // one TDM op moves the whole 64x64 bf16 A tile
      const unsigned long long ga =
          (unsigned long long)(uintptr_t)(a_in + (size_t)m0 * DIM + k0);
      u32x4 g0;
      g0.x = 1u;                                         // D# count = 1
      g0.y = (unsigned)(uintptr_t)At;                    // lds_addr (bytes)
      g0.z = (unsigned)(ga & 0xffffffffu);               // global_addr[31:0]
      g0.w = (unsigned)((ga >> 32) & 0x01ffffffu) | 0x80000000u;  // type=2
      const i32x8 g1 = {
          0x00010000,   // data_size=1 (2B), no multicast/pad/iterate
          0x04000000,   // tensor_dim0 = 1024 (low16 @ bit48)
          0x20000000,   // tensor_dim1 = 8192 (low16 @ bit80)
          0x00400000,   // tile_dim0 = 64 (@ bit112)
          64,           // tile_dim1 = 64
          1024,         // tensor_dim0_stride = 1024 elements
          0, 0};
      const i32x4 z4 = {0, 0, 0, 0};
      const i32x8 z8 = {0, 0, 0, 0, 0, 0, 0, 0};
      // 6-arg form (clang-23 / therock headers): groups 2/3 + extra group zero
      __builtin_amdgcn_tensor_load_to_lds(g0, g1, z4, z4, z8, 0);
      __builtin_amdgcn_s_wait_tensorcnt(0);
    }
#else
    { // fallback: per-lane async DMA of the A tile
      const int row = tid >> 1, half = tid & 1;
      const __bf16* src = a_in + (size_t)(m0 + row) * DIM + k0 + 32 * half;
      const unsigned dst = (unsigned)(uintptr_t)(At + row * 64 + 32 * half);
      #pragma unroll
      for (int i = 0; i < 4; ++i)
        async_copy_b128(dst + 16 * i, (const char*)src + 16 * i);
    }
#endif
    { // stage raw fp32 B tile: contiguous async DMA (8 x 16B per thread)
      const int row = tid >> 1, half = tid & 1;
      const float* src = w + (size_t)(k0 + row) * DIM + n0 + 32 * half;
      const unsigned dst = (unsigned)(uintptr_t)(Bf + row * 64 + 32 * half);
      #pragma unroll
      for (int i = 0; i < 8; ++i)
        async_copy_b128(dst + 16 * i, (const char*)src + 16 * i);
    }
    wait_asynccnt0();
    __syncthreads();

    { // convert/transpose Bf[k][n] fp32 -> Bt[n][k] bf16 (conflict-free)
      const int n = tid & 63, kk = (tid >> 6) * 32;
      #pragma unroll
      for (int i = 0; i < 32; ++i)
        Bt[n * 64 + kk + i] = (__bf16)Bf[(size_t)(kk + i) * 64 + n];
    }
    __syncthreads();

    Frag16 a[2][2], b[2][2];
    #pragma unroll
    for (int mi = 0; mi < 2; ++mi)
      #pragma unroll
      for (int kc = 0; kc < 2; ++kc) {
        const __bf16* rowp = At + (wm + 16 * mi + l16) * 64 + 32 * kc;
        a[mi][kc].v = *(const v16bf*)(rowp + 16 * hl);
      }
    #pragma unroll
    for (int ni = 0; ni < 2; ++ni)
      #pragma unroll
      for (int kc = 0; kc < 2; ++kc)
        b[ni][kc].v =
            *(const v16bf*)(Bt + (wn + 16 * ni + l16) * 64 + 32 * kc + 16 * hl);
    #pragma unroll
    for (int mi = 0; mi < 2; ++mi)
      #pragma unroll
      for (int ni = 0; ni < 2; ++ni)
        #pragma unroll
        for (int kc = 0; kc < 2; ++kc)
          acc[mi][ni] = wmma_bf16(a[mi][kc].v, b[ni][kc].v, acc[mi][ni]);
    __syncthreads();
  }

  #pragma unroll
  for (int mi = 0; mi < 2; ++mi)
    #pragma unroll
    for (int ni = 0; ni < 2; ++ni)
      #pragma unroll
      for (int vr = 0; vr < 8; ++vr) {
        const int m = m0 + wm + 16 * mi + vr + 8 * hl;
        const int n = n0 + wn + 16 * ni + l16;
        out[(size_t)m * DIM + n] = acc[mi][ni][vr];
      }
}

// ---------------------------------------------------------------------------
extern "C" void kernel_launch(void* const* d_in, const int* in_sizes, int n_in,
                              void* d_out, int out_size, void* d_ws, size_t ws_size,
                              hipStream_t stream) {
  const float* x     = (const float*)d_in[0];
  // d_in[1] = causal mask: applied analytically in attn_kernel, unused here
  const float* w_qkv = (const float*)d_in[2];
  const float* w_out = (const float*)d_in[3];
  float* out = (float*)d_out;

  const size_t per = (size_t)BATCH * HEADS * SEQ * DHEAD;  // 8,388,608 elems
  __bf16* Qw  = (__bf16*)d_ws;
  __bf16* Kw  = Qw + per;
  __bf16* Vtw = Kw + per;
  __bf16* Aw  = Vtw + per;   // attention output [B,S,H,DH] bf16

  qkv_proj_kernel<<<dim3((8192 / 64) * ((3 * DIM) / 64)), dim3(128), 0, stream>>>(
      x, w_qkv, Qw, Kw, Vtw);

  attn_kernel<<<dim3((BATCH * HEADS * (SEQ / 32)) / 4), dim3(128), 0, stream>>>(
      Qw, Kw, Vtw, Aw);

  out_proj_kernel<<<dim3((8192 / 64) * (DIM / 64)), dim3(128), 0, stream>>>(
      Aw, w_out, out);
}
